// MyAttention_2370821947962
// MI455X (gfx1250) — matrix operand order repaired and checked by reading, not compiled, not run on hardware
//
#include <hip/hip_runtime.h>
#include <hip/hip_bf16.h>
#include <math.h>

// ---------------- problem constants (from setup_inputs) ----------------
#define BB   8
#define NN   16384
#define CC   64
#define HH   128
#define WW   128
#define SR   8
#define HW16 256          // (H/SR)*(W/SR)
#define KQ   9            // gaussian kernel size = SR+1
#define EPSF 1e-6f

typedef __attribute__((ext_vector_type(16))) __bf16 v16bf;
typedef __attribute__((ext_vector_type(8)))  float  v8f;

__device__ __forceinline__ v8f wmma_bf16(v16bf a, v16bf b, v8f c) {
    return __builtin_amdgcn_wmma_f32_16x16x32_bf16(false, a, false, b,
                                                   (short)0, c, false, false);
}

// Load a 16-element bf16 fragment from two contiguous 16-byte chunks.
// (slots 0..7 <- p0[0..7], slots 8..15 <- p1[0..7]) — two ds_load_b128.
__device__ __forceinline__ v16bf frag_cat(const __bf16* p0, const __bf16* p1) {
    union { uint4 u[2]; v16bf v; } U;
    U.u[0] = *(const uint4*)p0;
    U.u[1] = *(const uint4*)p1;
    return U.v;
}
// A-fragment (ISA 7.12.2 16-bit A 16x32): lane row m=L&15, chunks at
// base + 8g and base + 16 + 8g, base = row_ptr + koff.
__device__ __forceinline__ v16bf frag_a(const __bf16* rowbase, int g) {
    return frag_cat(rowbase + 8 * g, rowbase + 16 + 8 * g);
}
// B-fragment (32x16): lane col n=L&15, chunks at colbase + 8g and colbase + 16 + 8g.
__device__ __forceinline__ v16bf frag_b(const __bf16* colbase, int g) {
    return frag_cat(colbase + 8 * g, colbase + 16 + 8 * g);
}

// ======================= 0. zero scratch =======================
__global__ void tr_zero_kernel(float* __restrict__ p, long n) {
    long i = (long)blockIdx.x * blockDim.x + threadIdx.x;
    long stride = (long)gridDim.x * blockDim.x;
    for (; i < n; i += stride) p[i] = 0.0f;
}

// ======================= 1. token scatter (token2map accumulate) ===========
__global__ __launch_bounds__(256) void tr_scatter_kernel(
    const float* __restrict__ xs,     // B,N,C
    const float* __restrict__ loc,    // B,N,2
    const float* __restrict__ conf,   // B,N,1
    float* __restrict__ facc,         // B,H,W,C+1
    float* __restrict__ cacc)         // B,256,2
{
    long total = (long)BB * NN;
    long i = (long)blockIdx.x * blockDim.x + threadIdx.x;
    long stride = (long)gridDim.x * blockDim.x;
    for (; i < total; i += stride) {
        int b = (int)(i >> 14);
        float lx = fminf(fmaxf(loc[i * 2 + 0], -1.0f), 1.0f);
        float ly = fminf(fmaxf(loc[i * 2 + 1], -1.0f), 1.0f);
        int px = (int)rintf(0.5f * (lx + 1.0f) * WW - 0.5f);
        int py = (int)rintf(0.5f * (ly + 1.0f) * HH - 0.5f);
        px = min(max(px, 0), WW - 1);
        py = min(max(py, 0), HH - 1);
        size_t base = (((size_t)b * HH + py) * WW + px) * (size_t)(CC + 1);
        const float* src = xs + (size_t)i * CC;
        #pragma unroll 4
        for (int c = 0; c < CC; ++c) atomicAdd(&facc[base + c], src[c]);
        atomicAdd(&facc[base + CC], 1.0f);
        // conf map on the 16x16 grid (kernel_size == 1 path)
        int qx = (int)rintf(0.5f * (lx + 1.0f) * (WW / SR) - 0.5f);
        int qy = (int)rintf(0.5f * (ly + 1.0f) * (HH / SR) - 0.5f);
        qx = min(max(qx, 0), WW / SR - 1);
        qy = min(max(qy, 0), HH / SR - 1);
        size_t cb = ((size_t)b * HW16 + qy * (WW / SR) + qx) * 2;
        atomicAdd(&cacc[cb + 0], conf[i]);
        atomicAdd(&cacc[cb + 1], 1.0f);
    }
}

// ======================= 2. normalize feat in place, mask -> channel C =====
__global__ void tr_fnorm_kernel(float* __restrict__ facc) {
    long total = (long)BB * HH * WW;
    long p = (long)blockIdx.x * blockDim.x + threadIdx.x;
    long stride = (long)gridDim.x * blockDim.x;
    for (; p < total; p += stride) {
        size_t base = (size_t)p * (CC + 1);
        float cnt = facc[base + CC];
        float mask = (cnt > 0.0f) ? 1.0f : 0.0f;
        float inv = mask / (cnt + EPSF);
        for (int c = 0; c < CC; ++c) facc[base + c] *= inv;
        facc[base + CC] = mask;
    }
}

// ======================= 3. conf finalize =======================
__global__ void tr_conf_kernel(const float* __restrict__ cacc,
                               float* __restrict__ confm) {
    int i = blockIdx.x * blockDim.x + threadIdx.x;
    if (i < BB * HW16) {
        float s = cacc[2 * i], n = cacc[2 * i + 1];
        confm[i] = (n > 0.0f ? 1.0f : 0.0f) * s / (n + EPSF);
    }
}

// ======================= 4. gaussian hole fill (9x9, sigma=2, reflect) =====
__global__ __launch_bounds__(256) void tr_gfill_kernel(
    const float* __restrict__ facc,   // B,H,W,C+1 (normalized, mask in ch C)
    float* __restrict__ feat2)        // B,H,W,C filled
{
    long total = (long)BB * HH * WW * CC;
    long i = (long)blockIdx.x * blockDim.x + threadIdx.x;
    long stride = (long)gridDim.x * blockDim.x;
    const float inv2piv = 1.0f / (2.0f * 3.14159265358979323846f * 4.0f);
    for (; i < total; i += stride) {
        int c = (int)(i & (CC - 1));
        long p = i >> 6;
        int xc = (int)(p & (WW - 1));
        long p2 = p >> 7;
        int yc = (int)(p2 & (HH - 1));
        int b = (int)(p2 >> 7);
        const float* base = facc + (size_t)b * HH * WW * (CC + 1);
        size_t ctr = ((size_t)yc * WW + xc) * (CC + 1);
        float mask = base[ctr + CC];
        float fv = base[ctr + c];
        float outv;
        if (mask > 0.0f) {
            outv = fv;
        } else {
            float g1[KQ];
            #pragma unroll
            for (int j = 0; j < KQ; ++j) {
                float d = (float)j - 4.0f;
                g1[j] = __expf(-d * d * 0.125f);   // var = 4 -> /(2*var)=0.125
            }
            float yv = 0.0f, ym = 0.0f;
            for (int dy = 0; dy < KQ; ++dy) {
                int sy = yc + dy - 4;
                sy = (sy < 0) ? -sy : ((sy >= HH) ? 2 * HH - 2 - sy : sy);
                for (int dx = 0; dx < KQ; ++dx) {
                    int sx = xc + dx - 4;
                    sx = (sx < 0) ? -sx : ((sx >= WW) ? 2 * WW - 2 - sx : sx);
                    float w = g1[dy] * g1[dx] * inv2piv;
                    const float* pp = base + ((size_t)sy * WW + sx) * (CC + 1);
                    yv += w * pp[c];
                    ym += w * pp[CC];
                }
            }
            outv = (ym > 0.0f) ? (yv / (ym + EPSF)) : 0.0f;   // feat==0 here
        }
        feat2[i] = outv;
    }
}

// ======================= 5. 8x8 stride-8 conv (im2col K=4096) ==============
__global__ __launch_bounds__(256) void tr_srconv_kernel(
    const float* __restrict__ feat2,  // B,H,W,C
    const float* __restrict__ sr_w,   // 8,8,C,C  ((ky*8+kx)*64+ci)*64+co
    const float* __restrict__ sr_b,   // C
    float* __restrict__ xs_raw)       // B,256,C
{
    int blk = blockIdx.x;             // B*256 output pixels
    int b = blk >> 8;
    int pix = blk & 255;
    int oy = pix >> 4, ox = pix & 15;
    int t = threadIdx.x;
    int co = t & 63, part = t >> 6;   // 4-way K split
    float acc = 0.0f;
    for (int k = part; k < SR * SR * CC; k += 4) {
        int ci = k & 63;
        int sp = k >> 6;
        int kx = sp & 7, ky = sp >> 3;
        float fv = feat2[(((size_t)b * HH + oy * SR + ky) * WW + ox * SR + kx) * CC + ci];
        acc += fv * sr_w[(size_t)k * CC + co];
    }
    __shared__ float red[256];
    red[t] = acc;
    __syncthreads();
    if (part == 0)
        xs_raw[(size_t)blk * CC + co] =
            red[co] + red[co + 64] + red[co + 128] + red[co + 192] + sr_b[co];
}

// ======================= 6. fused LayerNorm + KV projection -> bf16 ========
// kbuf stored [b][j][d] (K rows, d-contiguous -> B-frag for scores).
// vbuf stored [b][d][j] (transposed -> B-frag for P@V is K-contiguous).
__global__ __launch_bounds__(64) void tr_lnkv_kernel(
    const float* __restrict__ xs_raw,   // B*256, C
    const float* __restrict__ norm_w,
    const float* __restrict__ norm_b,
    const float* __restrict__ kv_w,     // 2C, C
    __bf16* __restrict__ kbuf,          // B, 256, C
    __bf16* __restrict__ vbuf)          // B, C, 256 (transposed)
{
    __shared__ float xn[CC];
    __shared__ float red[CC];
    int row = blockIdx.x;               // B*256
    int b = row >> 8;
    int m = row & 255;
    int t = threadIdx.x;
    float xv = xs_raw[(size_t)row * CC + t];
    red[t] = xv;
    __syncthreads();
    for (int off = 32; off >= 1; off >>= 1) {
        if (t < off) red[t] += red[t + off];
        __syncthreads();
    }
    float mean = red[0] * (1.0f / CC);
    __syncthreads();
    float d = xv - mean;
    red[t] = d * d;
    __syncthreads();
    for (int off = 32; off >= 1; off >>= 1) {
        if (t < off) red[t] += red[t + off];
        __syncthreads();
    }
    float var = red[0] * (1.0f / CC);
    __syncthreads();
    xn[t] = d * rsqrtf(var + 1e-5f) * norm_w[t] + norm_b[t];
    __syncthreads();
    float a0 = 0.0f, a1 = 0.0f;
    #pragma unroll 8
    for (int c = 0; c < CC; ++c) {
        float xc = xn[c];
        a0 += xc * kv_w[(size_t)t * CC + c];          // k rows 0..63
        a1 += xc * kv_w[(size_t)(t + CC) * CC + c];   // v rows 64..127
    }
    kbuf[(size_t)row * CC + t] = (__bf16)a0;
    vbuf[((size_t)b * CC + t) * HW16 + m] = (__bf16)a1;
}

// ======================= 7. weight prep: bf16 copy ([n][k] layout) =========
// B[k][n] = W[n][k]; storing W row-major as [n][k] makes B-fragments
// K-contiguous per output column.
__global__ void tr_wprep_kernel(const float* __restrict__ q_w,
                                const float* __restrict__ proj_w,
                                __bf16* __restrict__ qwb,
                                __bf16* __restrict__ pwb) {
    int i = blockIdx.x * blockDim.x + threadIdx.x;   // 4096
    if (i < CC * CC) {
        qwb[i] = (__bf16)q_w[i];
        pwb[i] = (__bf16)proj_w[i];
    }
}

// ======================= 8. fused attention megakernel =====================
// q = x @ qW^T ; S = q k^T * scale + conf ; P = softmax(S) ; O = P v ;
// out = O @ projW^T + bias.  One wave = one 16-row query tile; 8 waves/block.
__global__ __launch_bounds__(256) void tr_attn_kernel(
    const float* __restrict__ x,        // B,N,C
    const __bf16* __restrict__ kg,      // B,256,C
    const __bf16* __restrict__ vg,      // B,C,256 (transposed)
    const float* __restrict__ confg,    // B,256
    const __bf16* __restrict__ qw,      // [n][k] 64x64
    const __bf16* __restrict__ pw,      // [n][k] 64x64
    const float* __restrict__ proj_b,   // 64
    float* __restrict__ out)            // B,N,C
{
    extern __shared__ char smem[];
    __bf16* lds_k   = (__bf16*)smem;                 // [j][d] 256x64
    __bf16* lds_vT  = lds_k + HW16 * CC;             // [d][j] 64x256
    __bf16* lds_qw  = lds_vT + CC * HW16;            // [n][k] 64x64
    __bf16* lds_pw  = lds_qw + CC * CC;              // [n][k] 64x64
    float*  lds_cf  = (float*)(lds_pw + CC * CC);    // 256
    __bf16* lds_stg = (__bf16*)(lds_cf + HW16);      // 8 waves * (16*256 + 16*64)

    int b  = blockIdx.x >> 7;            // 128 row-blocks per batch
    int rb = blockIdx.x & 127;
    int tid = threadIdx.x;

    // ---- cooperative b128 copies into LDS ----
    {
        const uint4* sk = (const uint4*)(kg + (size_t)b * HW16 * CC);
        const uint4* sv = (const uint4*)(vg + (size_t)b * CC * HW16);
        uint4* dk = (uint4*)lds_k;
        uint4* dv = (uint4*)lds_vT;
        #pragma unroll
        for (int i = tid; i < HW16 * CC / 8; i += 256) {
            dk[i] = sk[i];
            dv[i] = sv[i];
        }
        const uint4* sq = (const uint4*)qw;
        const uint4* sp = (const uint4*)pw;
        uint4* dq = (uint4*)lds_qw;
        uint4* dp = (uint4*)lds_pw;
        #pragma unroll
        for (int i = tid; i < CC * CC / 8; i += 256) {
            dq[i] = sq[i];
            dp[i] = sp[i];
        }
        if (tid < HW16) lds_cf[tid] = confg[(size_t)b * HW16 + tid];
    }
    __syncthreads();

    int wave = tid >> 5, lane = tid & 31;
    int ln = lane & 15, g = lane >> 4;
    int m0 = rb * 128 + wave * 16;
    __bf16* stp = lds_stg + wave * (16 * HW16 + 16 * CC);  // P tile 16x256
    __bf16* sts = stp + 16 * HW16;                         // small tile 16x64

    // ---- load x row as A fragments (K=64 -> 2 chunks), via float4 ----
    const float* xrow = x + ((size_t)b * NN + m0 + ln) * CC;
    __builtin_prefetch(xrow, 0, 3);
    v16bf ax0, ax1;
    {
        float4 c0 = *(const float4*)(xrow + 8 * g);
        float4 c1 = *(const float4*)(xrow + 8 * g + 4);
        float4 c2 = *(const float4*)(xrow + 16 + 8 * g);
        float4 c3 = *(const float4*)(xrow + 16 + 8 * g + 4);
        float4 c4 = *(const float4*)(xrow + 32 + 8 * g);
        float4 c5 = *(const float4*)(xrow + 32 + 8 * g + 4);
        float4 c6 = *(const float4*)(xrow + 48 + 8 * g);
        float4 c7 = *(const float4*)(xrow + 48 + 8 * g + 4);
        ax0[0] = (__bf16)c0.x; ax0[1] = (__bf16)c0.y; ax0[2] = (__bf16)c0.z; ax0[3] = (__bf16)c0.w;
        ax0[4] = (__bf16)c1.x; ax0[5] = (__bf16)c1.y; ax0[6] = (__bf16)c1.z; ax0[7] = (__bf16)c1.w;
        ax0[8] = (__bf16)c2.x; ax0[9] = (__bf16)c2.y; ax0[10] = (__bf16)c2.z; ax0[11] = (__bf16)c2.w;
        ax0[12] = (__bf16)c3.x; ax0[13] = (__bf16)c3.y; ax0[14] = (__bf16)c3.z; ax0[15] = (__bf16)c3.w;
        ax1[0] = (__bf16)c4.x; ax1[1] = (__bf16)c4.y; ax1[2] = (__bf16)c4.z; ax1[3] = (__bf16)c4.w;
        ax1[4] = (__bf16)c5.x; ax1[5] = (__bf16)c5.y; ax1[6] = (__bf16)c5.z; ax1[7] = (__bf16)c5.w;
        ax1[8] = (__bf16)c6.x; ax1[9] = (__bf16)c6.y; ax1[10] = (__bf16)c6.z; ax1[11] = (__bf16)c6.w;
        ax1[12] = (__bf16)c7.x; ax1[13] = (__bf16)c7.y; ax1[14] = (__bf16)c7.z; ax1[15] = (__bf16)c7.w;
    }

    // ---- q = x @ qW^T (4 n-tiles x 2 k-steps), stage as bf16 ----
    #pragma unroll
    for (int n0 = 0; n0 < CC; n0 += 16) {
        const __bf16* col = lds_qw + (n0 + ln) * CC;      // [n][k]
        v16bf b0 = frag_b(col, g);
        v16bf b1 = frag_b(col + 32, g);
        v8f acc = {};
        acc = wmma_bf16(ax0, b0, acc);
        acc = wmma_bf16(ax1, b1, acc);
        #pragma unroll
        for (int r = 0; r < 8; ++r)
            sts[(r + 8 * g) * CC + n0 + ln] = (__bf16)acc[r];
    }
    __syncthreads();

    // ---- q tile as A fragments (2 x ds_load_b128 each) ----
    v16bf aq0 = frag_a(sts + ln * CC, g);
    v16bf aq1 = frag_a(sts + ln * CC + 32, g);

    // ---- scores: 16 key tiles, each 2 k-steps ----
    v8f Sv[16];
    #pragma unroll
    for (int t = 0; t < 16; ++t) {
        const __bf16* krow = lds_k + (t * 16 + ln) * CC;  // B[d][j] = k[j][d]
        v16bf bk0 = frag_b(krow, g);
        v16bf bk1 = frag_b(krow + 32, g);
        v8f acc = {};
        acc = wmma_bf16(aq0, bk0, acc);
        acc = wmma_bf16(aq1, bk1, acc);
        Sv[t] = acc;
    }

    // ---- scale + conf bias + softmax over 256 cols, stage P as bf16 ----
    const float scale = 0.125f;   // 64^-0.5
    float cfv[16];
    #pragma unroll
    for (int t = 0; t < 16; ++t) cfv[t] = lds_cf[ln + 16 * t];
    #pragma unroll
    for (int r = 0; r < 8; ++r) {
        float mx = -1e30f;
        #pragma unroll
        for (int t = 0; t < 16; ++t) {
            float sv = Sv[t][r] * scale + cfv[t];
            Sv[t][r] = sv;
            mx = fmaxf(mx, sv);
        }
        for (int off = 1; off < 16; off <<= 1)
            mx = fmaxf(mx, __shfl_xor(mx, off, 32));
        float sum = 0.0f;
        #pragma unroll
        for (int t = 0; t < 16; ++t) {
            float e = __expf(Sv[t][r] - mx);
            Sv[t][r] = e;
            sum += e;
        }
        for (int off = 1; off < 16; off <<= 1)
            sum += __shfl_xor(sum, off, 32);
        float inv = 1.0f / sum;
        #pragma unroll
        for (int t = 0; t < 16; ++t)
            stp[(r + 8 * g) * HW16 + ln + 16 * t] = (__bf16)(Sv[t][r] * inv);
    }
    __syncthreads();

    // ---- preload the 8 A-fragments of P (K=256) once ----
    v16bf ap[8];
    #pragma unroll
    for (int kb = 0; kb < 8; ++kb)
        ap[kb] = frag_a(stp + ln * HW16 + kb * 32, g);

    // ---- O = P @ v : per 16-wide d tile, 8 k-steps ----
    #pragma unroll
    for (int d0 = 0; d0 < CC; d0 += 16) {
        const __bf16* vrow = lds_vT + (d0 + ln) * HW16;   // B[j][d] = vT[d][j]
        v8f acc = {};
        #pragma unroll
        for (int kb = 0; kb < 8; ++kb)
            acc = wmma_bf16(ap[kb], frag_b(vrow + kb * 32, g), acc);
        #pragma unroll
        for (int r = 0; r < 8; ++r)
            sts[(r + 8 * g) * CC + d0 + ln] = (__bf16)acc[r];
    }
    __syncthreads();

    // ---- out = O @ projW^T + bias ----
    v16bf ao0 = frag_a(sts + ln * CC, g);
    v16bf ao1 = frag_a(sts + ln * CC + 32, g);
    #pragma unroll
    for (int n0 = 0; n0 < CC; n0 += 16) {
        const __bf16* col = lds_pw + (n0 + ln) * CC;
        v16bf b0 = frag_b(col, g);
        v16bf b1 = frag_b(col + 32, g);
        v8f acc = {};
        acc = wmma_bf16(ao0, b0, acc);
        acc = wmma_bf16(ao1, b1, acc);
        float bias = proj_b[n0 + ln];
        #pragma unroll
        for (int r = 0; r < 8; ++r)
            out[((size_t)b * NN + m0 + r + 8 * g) * CC + n0 + ln] = acc[r] + bias;
    }
}

// ======================= host: launch sequence =======================
extern "C" void kernel_launch(void* const* d_in, const int* in_sizes, int n_in,
                              void* d_out, int out_size, void* d_ws, size_t ws_size,
                              hipStream_t stream) {
    const float* x        = (const float*)d_in[0];
    const float* x_source = (const float*)d_in[1];
    const float* loc      = (const float*)d_in[2];
    const float* conf_src = (const float*)d_in[3];
    const float* q_w      = (const float*)d_in[4];
    const float* kv_w     = (const float*)d_in[5];
    const float* sr_w     = (const float*)d_in[6];
    const float* sr_b     = (const float*)d_in[7];
    const float* norm_w   = (const float*)d_in[8];
    const float* norm_b   = (const float*)d_in[9];
    const float* proj_w   = (const float*)d_in[10];
    const float* proj_b   = (const float*)d_in[11];
    float* out = (float*)d_out;

    // ---- workspace layout ----
    char* ws = (char*)d_ws;
    size_t off = 0;
    auto alloc = [&](size_t bytes) {
        char* p = ws + off;
        off += (bytes + 255) & ~(size_t)255;
        return p;
    };
    float*  facc   = (float*)alloc((size_t)BB * HH * WW * (CC + 1) * 4); // 34MB
    float*  cacc   = (float*)alloc((size_t)BB * HW16 * 2 * 4);
    float*  feat2  = (float*)alloc((size_t)BB * HH * WW * CC * 4);       // 32MB
    float*  confm  = (float*)alloc((size_t)BB * HW16 * 4);
    float*  xs_raw = (float*)alloc((size_t)BB * HW16 * CC * 4);
    __bf16* kbuf   = (__bf16*)alloc((size_t)BB * HW16 * CC * 2);
    __bf16* vbuf   = (__bf16*)alloc((size_t)BB * CC * HW16 * 2);
    __bf16* qwb    = (__bf16*)alloc((size_t)CC * CC * 2);
    __bf16* pwb    = (__bf16*)alloc((size_t)CC * CC * 2);

    // 0. zero the accumulators
    long nfacc = (long)BB * HH * WW * (CC + 1);
    tr_zero_kernel<<<2048, 256, 0, stream>>>(facc, nfacc);
    tr_zero_kernel<<<8, 256, 0, stream>>>(cacc, (long)BB * HW16 * 2);

    // 1. scatter tokens
    tr_scatter_kernel<<<512, 256, 0, stream>>>(x_source, loc, conf_src, facc, cacc);

    // 2. normalize feat + mask
    tr_fnorm_kernel<<<512, 256, 0, stream>>>(facc);

    // 3. conf map finalize
    tr_conf_kernel<<<(BB * HW16 + 255) / 256, 256, 0, stream>>>(cacc, confm);

    // 4. gaussian hole fill
    tr_gfill_kernel<<<4096, 256, 0, stream>>>(facc, feat2);

    // 5. 8x8/8 conv
    tr_srconv_kernel<<<BB * HW16, 256, 0, stream>>>(feat2, sr_w, sr_b, xs_raw);

    // 6. fused LN + KV projection (bf16 K/V out; V transposed)
    tr_lnkv_kernel<<<BB * HW16, 64, 0, stream>>>(xs_raw, norm_w, norm_b, kv_w,
                                                 kbuf, vbuf);

    // 7. weight copy -> bf16 ([n][k] layout)
    tr_wprep_kernel<<<(CC * CC + 255) / 256, 256, 0, stream>>>(q_w, proj_w, qwb, pwb);

    // 8. fused q-proj + attention + out-proj
    size_t lds_bytes =
        (size_t)(2 * HW16 * CC + 2 * CC * CC) * sizeof(__bf16) +   // K,V,Wq,Wp
        (size_t)HW16 * sizeof(float) +                              // conf
        (size_t)8 * (16 * HW16 + 16 * CC) * sizeof(__bf16);         // staging
    tr_attn_kernel<<<BB * 128, 256, lds_bytes, stream>>>(
        x, kbuf, vbuf, confm, qwb, pwb, proj_b, out);

    (void)in_sizes; (void)n_in; (void)out_size; (void)ws_size;
}
